// GIMM_43765716746932
// MI455X (gfx1250) — matrix operand order repaired and checked by reading, not compiled, not run on hardware
//
#include <hip/hip_runtime.h>
#include <hip/hip_bf16.h>

typedef __attribute__((ext_vector_type(16))) _Float16 v16h;
typedef __attribute__((ext_vector_type(8)))  float    v8f;

static constexpr int BB = 4;
static constexpr int HH = 384;
static constexpr int WW = 640;
static constexpr int HW = HH * WW;

// ---------------------------------------------------------------- helpers
__device__ __forceinline__ v16h ld_frag16(const void* p) {
  union { uint4 q[2]; v16h h; } u;
  u.q[0] = ((const uint4*)p)[0];
  u.q[1] = ((const uint4*)p)[1];
  return u.h;
}
__device__ __forceinline__ float lreluf(float v) { return v >= 0.f ? v : 0.1f * v; }
__device__ __forceinline__ int refl(int i, int n) { if (i < 0) i = -i; if (i >= n) i = 2*n - 2 - i; return i; }

// ============================================================ weight packing
// Pack conv weights (OIHW f32) into WMMA A-fragment order (16x32 f16 tiles).
// Global K index: k = tap*CIN + cin, zero padded to CHUNKS*32.
// A layout: lane l -> M = l%16, half = l/16; vgpr j: K = half*16 + 2j (+1).
__global__ void prep_conv_frags(const float* __restrict__ w, unsigned int* __restrict__ frag,
                                int cout, int cin) {
  int chunks = (cin * 9 + 31) / 32;
  int total = (cout / 16) * chunks * 256;
  int i = blockIdx.x * blockDim.x + threadIdx.x;
  if (i >= total) return;
  int j = i & 7, lane = (i >> 3) & 31, tc = i >> 8;
  int c = tc % chunks, t = tc / chunks;
  int half = lane >> 4, m = t * 16 + (lane & 15);
  int kloc = half * 16 + 2 * j;
  unsigned int pack = 0;
#pragma unroll
  for (int e = 0; e < 2; ++e) {
    int k = c * 32 + kloc + e;
    float v = 0.f;
    if (k < cin * 9) { int tap = k / cin, ci = k % cin; v = w[((size_t)m * cin + ci) * 9 + tap]; }
    union { _Float16 h; unsigned short u; } cv; cv.h = (_Float16)v;
    pack |= (unsigned int)cv.u << (16 * e);
  }
  frag[i] = pack;
}

// Pack MLP weights W (in_dim x out_dim, row-major) as A = W^T, K padded to kpad.
__global__ void prep_mlp_frags(const float* __restrict__ w, unsigned int* __restrict__ frag,
                               int in_dim, int out_dim, int kpad) {
  int chunks = kpad / 32;
  int total = (out_dim / 16) * chunks * 256;
  int i = blockIdx.x * blockDim.x + threadIdx.x;
  if (i >= total) return;
  int j = i & 7, lane = (i >> 3) & 31, tc = i >> 8;
  int c = tc % chunks, t = tc / chunks;
  int half = lane >> 4, m = t * 16 + (lane & 15);
  int kloc = half * 16 + 2 * j;
  unsigned int pack = 0;
#pragma unroll
  for (int e = 0; e < 2; ++e) {
    int k = c * 32 + kloc + e;
    float v = (k < in_dim) ? w[(size_t)k * out_dim + m] : 0.f;
    union { _Float16 h; unsigned short u; } cv; cv.h = (_Float16)v;
    pack |= (unsigned int)cv.u << (16 * e);
  }
  frag[i] = pack;
}

// ============================================================ WMMA conv3x3
// One block: output row gy, 128 pixels, all COUT channels. 8 waves x 16px.
template<int CIN, int COUT, int PROWS>
__global__ __launch_bounds__(256) void conv3x3_wmma(
    const _Float16* __restrict__ in, int in_cs,
    const unsigned int* __restrict__ frag,
    const float* __restrict__ bias,
    _Float16* __restrict__ out, int out_cs,
    const _Float16* __restrict__ res, int res_cs,
    int inact, int outact, int reflect) {
  constexpr int TW = 128;
  constexpr int CP = CIN + 8;                 // padded cin stride (f16)
  constexpr int CHUNKS = (CIN * 9 + 31) / 32;
  constexpr int TILES = COUT / 16;
  __shared__ alignas(16) _Float16 patch[PROWS * (TW + 2) * CP];

  const int tid = threadIdx.x;
  const int b = blockIdx.z, gy = blockIdx.y, x0 = blockIdx.x * TW;

  if (PROWS == 4) {                           // zero pad-tap row (CIN==16 K padding)
    for (int i = tid; i < (TW + 2) * CP; i += 256)
      patch[3 * (TW + 2) * CP + i] = (_Float16)0.f;
  }
  for (int i = tid; i < 3 * (TW + 2) * CIN; i += 256) {
    int x = i % (TW + 2);
    int r = (i / (TW + 2)) % 3;
    int c = i / ((TW + 2) * 3);
    int yy = gy + r - 1, xx = x0 + x - 1;
    float v;
    if (reflect) {
      yy = refl(yy, HH); xx = refl(xx, WW);
      v = (float)in[((size_t)b * in_cs + c) * HW + (size_t)yy * WW + xx];
    } else {
      v = (yy < 0 || yy >= HH || xx < 0 || xx >= WW) ? 0.f
          : (float)in[((size_t)b * in_cs + c) * HW + (size_t)yy * WW + xx];
    }
    if (inact) v = lreluf(v);
    patch[(r * (TW + 2) + x) * CP + c] = (_Float16)v;
  }
  __syncthreads();

  const int lane = tid & 31, wave = tid >> 5;
  const int half = lane >> 4, nl = lane & 15;
  const int n0 = wave * 16;

  v8f acc[TILES];
#pragma unroll
  for (int t = 0; t < TILES; ++t)
#pragma unroll
    for (int r = 0; r < 8; ++r) acc[t][r] = 0.f;

  for (int c = 0; c < CHUNKS; ++c) {
    const int kb = c * 32 + half * 16;        // 16 contiguous K per lane-half
    const int tap = kb / CIN, cin0 = kb % CIN;
    const int dy = tap / 3, dx = tap - dy * 3;
    v16h bfrag = ld_frag16(&patch[(dy * (TW + 2) + (n0 + nl + dx)) * CP + cin0]);
#pragma unroll
    for (int t = 0; t < TILES; ++t) {
      v16h afrag = ld_frag16(frag + (size_t)((t * CHUNKS + c) * 32 + lane) * 8);
      acc[t] = __builtin_amdgcn_wmma_f32_16x16x32_f16(
          false, afrag, false, bfrag, (short)0, acc[t], false, false);
    }
  }

  const int gx = x0 + n0 + nl;                // C/D layout: N=nl, M = t*16+half*8+r
#pragma unroll
  for (int t = 0; t < TILES; ++t) {
#pragma unroll
    for (int r = 0; r < 8; ++r) {
      const int m = t * 16 + half * 8 + r;
      float v = acc[t][r] + bias[m];
      if (outact) v = lreluf(v);
      if (res) v += (float)res[((size_t)b * res_cs + m) * HW + (size_t)gy * WW + gx];
      out[((size_t)b * out_cs + m) * HW + (size_t)gy * WW + gx] = (_Float16)v;
    }
  }
}

// ============================================================ enc c0 (2->16, scalar)
__global__ __launch_bounds__(256) void enc_c0_kernel(const float* __restrict__ xs, int frame,
                                                     const float* __restrict__ w,
                                                     const float* __restrict__ bias,
                                                     _Float16* __restrict__ out) {
  __shared__ float ws_[288 + 16];
  for (int i = threadIdx.x; i < 304; i += 256) ws_[i] = (i < 288) ? w[i] : bias[i - 288];
  __syncthreads();
  int gid = blockIdx.x * 256 + threadIdx.x;
  if (gid >= BB * HW) return;
  int b = gid / HW, p = gid - b * HW;
  int y = p / WW, x = p - y * WW;
  float patch[2][9];
#pragma unroll
  for (int ci = 0; ci < 2; ++ci)
#pragma unroll
    for (int t = 0; t < 9; ++t) {
      int yy = y + t / 3 - 1, xx = x + t % 3 - 1;
      patch[ci][t] = (yy < 0 || yy >= HH || xx < 0 || xx >= WW) ? 0.f
                     : xs[(((size_t)b * 2 + ci) * 2 + frame) * HW + (size_t)yy * WW + xx];
    }
  for (int co = 0; co < 16; ++co) {
    float a = ws_[288 + co];
#pragma unroll
    for (int ci = 0; ci < 2; ++ci)
#pragma unroll
      for (int t = 0; t < 9; ++t) a += ws_[(co * 2 + ci) * 9 + t] * patch[ci][t];
    out[((size_t)b * 16 + co) * HW + p] = (_Float16)a;
  }
}

// ============================================================ splat weights
__device__ __forceinline__ float2 bisample2(const float* px, const float* py, float sx, float sy) {
  float x0 = floorf(sx), y0 = floorf(sy);
  float rx = 0.f, ry = 0.f;
#pragma unroll
  for (int d = 0; d < 4; ++d) {
    float xi = x0 + (d & 1), yi = y0 + (d >> 1);
    float wg = (1.f - fabsf(sx - xi)) * (1.f - fabsf(sy - yi));
    if (xi >= 0.f && xi <= (float)(WW - 1) && yi >= 0.f && yi <= (float)(HH - 1)) {
      int idx = (int)yi * WW + (int)xi;
      rx += px[idx] * wg; ry += py[idx] * wg;
    }
  }
  return make_float2(rx, ry);
}

__global__ void splat_weights_kernel(const float* __restrict__ fl, const float* __restrict__ g,
                                     const float* __restrict__ afe, const float* __restrict__ av,
                                     float* __restrict__ w1, float* __restrict__ w2) {
  int gid = blockIdx.x * blockDim.x + threadIdx.x;
  if (gid >= BB * HW) return;
  int b = gid / HW, p = gid - b * HW;
  int y = p / WW, x = p - y * WW;
  // ori_flow layout (B,2comp,2dir,H,W): plane ((b*2+c)*2+s)
  const float* F01x = fl + (size_t)(b * 4 + 0) * HW;
  const float* F10x = fl + (size_t)(b * 4 + 1) * HW;
  const float* F01y = fl + (size_t)(b * 4 + 2) * HW;
  const float* F10y = fl + (size_t)(b * 4 + 3) * HW;
  float s01x = 0, s01y = 0, q01x = 0, q01y = 0;
  float s10x = 0, s10y = 0, q10x = 0, q10y = 0;
#pragma unroll
  for (int t = 0; t < 9; ++t) {
    int yy = refl(y + t / 3 - 1, HH), xx = refl(x + t % 3 - 1, WW);
    float gw = g[t];
    int idx = yy * WW + xx;
    float a = F01x[idx], bv = F01y[idx], c = F10x[idx], d = F10y[idx];
    s01x += gw * a; q01x += gw * a * a; s01y += gw * bv; q01y += gw * bv * bv;
    s10x += gw * c; q10x += gw * c * c; s10y += gw * d; q10y += gw * d * d;
  }
  float var01 = 0.5f * (sqrtf(fmaxf(q01x - s01x * s01x, 1e-9f)) + sqrtf(fmaxf(q01y - s01y * s01y, 1e-9f)));
  float var10 = 0.5f * (sqrtf(fmaxf(q10x - s10x * s10x, 1e-9f)) + sqrtf(fmaxf(q10y - s10y * s10y, 1e-9f)));
  int p0 = y * WW + x;
  float f01x = F01x[p0], f01y = F01y[p0], f10x = F10x[p0], f10y = F10y[p0];
  float2 wA = bisample2(F10x, F10y, (float)x + f01x, (float)y + f01y);   // f01_warp = -warp(f10,f01)
  float2 wB = bisample2(F01x, F01y, (float)x + f10x, (float)y + f10y);
  float err01 = 0.5f * (fabsf(-wA.x - f01x) + fabsf(-wA.y - f01y));
  float err10 = 0.5f * (fabsf(-wB.x - f10x) + fabsf(-wB.y - f10y));
  float fe = afe[0], vv = av[0];
  w1[gid] = 1.f / (1.f + err01 * fe) + 1.f / (1.f + var01 * vv);
  w2[gid] = 1.f / (1.f + err10 * fe) + 1.f / (1.f + var10 * vv);
}

// ============================================================ splat / normalize
__global__ void zero_f32(float* __restrict__ p, int n) {
  for (int i = blockIdx.x * blockDim.x + threadIdx.x; i < n; i += gridDim.x * blockDim.x) p[i] = 0.f;
}

__global__ void splat_kernel(const _Float16* __restrict__ pl, int cs,
                             const float* __restrict__ fl, int s,
                             const float* __restrict__ ts,
                             const float* __restrict__ metric,
                             float* __restrict__ num, float* __restrict__ den) {
  int gid = blockIdx.x * blockDim.x + threadIdx.x;
  if (gid >= BB * HW) return;
  int b = gid / HW, p = gid - b * HW;
  int y = p / WW, x = p - y * WW;
  float t = ts[b];
  float sc = (s == 0) ? t : (1.f - t);
  float fx = fl[(size_t)(b * 4 + 0 * 2 + s) * HW + p] * sc;
  float fy = fl[(size_t)(b * 4 + 1 * 2 + s) * HW + p] * sc;
  float wexp = expf(metric[gid]);
  float vals[16];
#pragma unroll
  for (int c = 0; c < 16; ++c) vals[c] = (float)pl[((size_t)b * cs + c) * HW + p] * wexp;
  float sx = (float)x + fx, sy = (float)y + fy;
  float x0 = floorf(sx), y0 = floorf(sy);
#pragma unroll
  for (int d = 0; d < 4; ++d) {
    float xi = x0 + (d & 1), yi = y0 + (d >> 1);
    float wg = (1.f - fabsf(sx - xi)) * (1.f - fabsf(sy - yi));
    if (xi >= 0.f && xi <= (float)(WW - 1) && yi >= 0.f && yi <= (float)(HH - 1) && wg != 0.f) {
      int idx = (int)yi * WW + (int)xi;
      for (int c = 0; c < 16; ++c)
        atomicAdd(&num[((size_t)b * 16 + c) * HW + idx], vals[c] * wg);
      atomicAdd(&den[(size_t)b * HW + idx], wexp * wg);
    }
  }
}

__global__ void normalize_kernel(const float* __restrict__ num, const float* __restrict__ den,
                                 _Float16* __restrict__ out, int out_cs) {
  int gid = blockIdx.x * blockDim.x + threadIdx.x;
  if (gid >= BB * 16 * HW) return;
  int p = gid % HW, c = (gid / HW) % 16, b = gid / (16 * HW);
  float d = den[(size_t)b * HW + p];
  d = (d == 0.f) ? 1.f : d;
  out[((size_t)b * out_cs + c) * HW + p] = (_Float16)(num[gid] / d);
}

// ============================================================ hyponet (fused MLP)
__global__ __launch_bounds__(256) void hyponet_kernel(
    const float* __restrict__ coord, const _Float16* __restrict__ lat,
    const unsigned int* __restrict__ frag1, const float* __restrict__ b1,
    const unsigned int* __restrict__ frag2, const float* __restrict__ b2,
    const float* __restrict__ w3, const float* __restrict__ b3,
    float* __restrict__ out) {
  __shared__ alignas(16) _Float16 h1[128 * 96];
  __shared__ alignas(16) _Float16 h2[128 * 64];
  __shared__ float w3s[130];
  const int tid = threadIdx.x;
  const int b = blockIdx.z, gy = blockIdx.y, x0 = blockIdx.x * 128;
  if (tid < 130) w3s[tid] = (tid < 128) ? w3[tid] : b3[tid - 128];
  if (tid < 128) {
    int p = tid, gx = x0 + p;
    size_t ci = ((size_t)b * HW + (size_t)gy * WW + gx) * 3;
    float tc = coord[ci], xc = coord[ci + 1], yc = coord[ci + 2];
    float f = 3.14159265358979323846f;        // pi * 2^fq
#pragma unroll
    for (int fq = 0; fq < 8; ++fq) {
      float st, ct, sx, cx, sy, cy;
      __sincosf(tc * f, &st, &ct);
      __sincosf(xc * f, &sx, &cx);
      __sincosf(yc * f, &sy, &cy);
      _Float16* hp = &h1[p * 96 + fq * 6];
      hp[0] = (_Float16)st; hp[1] = (_Float16)sx; hp[2] = (_Float16)sy;
      hp[3] = (_Float16)ct; hp[4] = (_Float16)cx; hp[5] = (_Float16)cy;
      f *= 2.f;
    }
    for (int c = 0; c < 32; ++c)
      h1[p * 96 + 48 + c] = lat[((size_t)b * 32 + c) * HW + (size_t)gy * WW + gx];
    for (int c = 80; c < 96; ++c) h1[p * 96 + c] = (_Float16)0.f;
  }
  __syncthreads();

  const int lane = tid & 31, wave = tid >> 5;
  const int half = lane >> 4, nl = lane & 15;
  const int p = wave * 16 + nl;

  v8f acc[4];
  // ---- layer 1: K=96 (3 chunks), M=64 (4 tiles)
#pragma unroll
  for (int t = 0; t < 4; ++t)
#pragma unroll
    for (int r = 0; r < 8; ++r) acc[t][r] = 0.f;
  for (int c = 0; c < 3; ++c) {
    v16h bf = ld_frag16(&h1[p * 96 + c * 32 + half * 16]);
#pragma unroll
    for (int t = 0; t < 4; ++t) {
      v16h af = ld_frag16(frag1 + (size_t)((t * 3 + c) * 32 + lane) * 8);
      acc[t] = __builtin_amdgcn_wmma_f32_16x16x32_f16(false, af, false, bf, (short)0, acc[t], false, false);
    }
  }
#pragma unroll
  for (int t = 0; t < 4; ++t)
#pragma unroll
    for (int r = 0; r < 8; ++r) {
      int m = t * 16 + half * 8 + r;
      float v = acc[t][r] + b1[m];
      h2[p * 64 + m] = (_Float16)(v > 0.f ? v : 0.f);
    }
  __syncthreads();
  // ---- layer 2: K=64 (2 chunks) -> write back into h1 region as [p][64]
#pragma unroll
  for (int t = 0; t < 4; ++t)
#pragma unroll
    for (int r = 0; r < 8; ++r) acc[t][r] = 0.f;
  for (int c = 0; c < 2; ++c) {
    v16h bf = ld_frag16(&h2[p * 64 + c * 32 + half * 16]);
#pragma unroll
    for (int t = 0; t < 4; ++t) {
      v16h af = ld_frag16(frag2 + (size_t)((t * 2 + c) * 32 + lane) * 8);
      acc[t] = __builtin_amdgcn_wmma_f32_16x16x32_f16(false, af, false, bf, (short)0, acc[t], false, false);
    }
  }
#pragma unroll
  for (int t = 0; t < 4; ++t)
#pragma unroll
    for (int r = 0; r < 8; ++r) {
      int m = t * 16 + half * 8 + r;
      float v = acc[t][r] + b2[m];
      h1[p * 64 + m] = (_Float16)(v > 0.f ? v : 0.f);
    }
  __syncthreads();
  // ---- layer 3 (64 -> 2) scalar
  {
    int pp = tid >> 1, o = tid & 1;
    float s = w3s[128 + o];
    for (int i = 0; i < 64; ++i) s += (float)h1[pp * 64 + i] * w3s[i * 2 + o];
    int gx = x0 + pp;
    out[((size_t)b * 2 + o) * HW + (size_t)gy * WW + gx] = s;   // (B,2,1,H,W)
  }
}

// ============================================================ host
static inline size_t fpc(int cout, int cin) { return (size_t)(cout / 16) * ((cin * 9 + 31) / 32) * 256; }
static inline size_t fpm(int outd, int kpad) { return (size_t)(outd / 16) * (kpad / 32) * 256; }

extern "C" void kernel_launch(void* const* d_in, const int* in_sizes, int n_in,
                              void* d_out, int out_size, void* d_ws, size_t ws_size,
                              hipStream_t stream) {
  (void)in_sizes; (void)n_in; (void)out_size; (void)ws_size;
  // Input order: setup_inputs insertion order; params flattened as jax pytree
  // (dict keys sorted, list/tuple order preserved):
  const float* xs      = (const float*)d_in[0];
  const float* oflow   = (const float*)d_in[1];
  const float* tsteps  = (const float*)d_in[2];
  const float* coord   = (const float*)d_in[3];
  const float* a_fe    = (const float*)d_in[4];
  const float* a_v     = (const float*)d_in[5];
  const float* ec0w = (const float*)d_in[6],  *ec0b = (const float*)d_in[7];
  const float* ec1w = (const float*)d_in[8],  *ec1b = (const float*)d_in[9];
  const float* ec2w = (const float*)d_in[10], *ec2b = (const float*)d_in[11];
  const float* lb1[3], *lb2[3], *lw1[3], *lw2[3];
  for (int l = 0; l < 3; ++l) {
    lb1[l] = (const float*)d_in[12 + l * 4 + 0];
    lb2[l] = (const float*)d_in[12 + l * 4 + 1];
    lw1[l] = (const float*)d_in[12 + l * 4 + 2];
    lw2[l] = (const float*)d_in[12 + l * 4 + 3];
  }
  const float* g    = (const float*)d_in[24];
  const float* h0w  = (const float*)d_in[25], *h0b = (const float*)d_in[26];
  const float* h1w  = (const float*)d_in[27], *h1b = (const float*)d_in[28];
  const float* h2w  = (const float*)d_in[29], *h2b = (const float*)d_in[30];
  const float* rc0w = (const float*)d_in[31], *rc0b = (const float*)d_in[32];
  const float* rc1w = (const float*)d_in[33], *rc1b = (const float*)d_in[34];
  const float* rc2w = (const float*)d_in[35], *rc2b = (const float*)d_in[36];
  const float* rlb1 = (const float*)d_in[37], *rlb2 = (const float*)d_in[38];
  const float* rlw1 = (const float*)d_in[39], *rlw2 = (const float*)d_in[40];

  // -------- workspace carve (time-aliased regions)
  char* ws = (char*)d_ws;
  size_t off = 0;
  auto take = [&](size_t bytes) -> char* {
    char* r = ws + off; off = (off + bytes + 255) & ~(size_t)255; return r;
  };
  _Float16* ping  = (_Float16*)take((size_t)BB * 32 * HW * 2);
  _Float16* act64 = (_Float16*)take((size_t)BB * 64 * HW * 2);
  _Float16* b64b  = (_Float16*)take((size_t)BB * 64 * HW * 2);  // alias: enc a0
  _Float16* b64c  = (_Float16*)take((size_t)BB * 64 * HW * 2);  // alias: pong, num/den
  float* w1 = (float*)take((size_t)BB * HW * 4);
  float* w2 = (float*)take((size_t)BB * HW * 4);
  unsigned int* f_ec1 = (unsigned int*)take(fpc(32, 16) * 4);
  unsigned int* f_lat[6];
  for (int i = 0; i < 6; ++i) f_lat[i] = (unsigned int*)take(fpc(32, 32) * 4);
  unsigned int* f_ec2  = (unsigned int*)take(fpc(16, 32) * 4);
  unsigned int* f_rc0  = (unsigned int*)take(fpc(32, 64) * 4);
  unsigned int* f_rc1  = (unsigned int*)take(fpc(64, 32) * 4);
  unsigned int* f_rl1  = (unsigned int*)take(fpc(64, 64) * 4);
  unsigned int* f_rl2  = (unsigned int*)take(fpc(64, 64) * 4);
  unsigned int* f_rc2  = (unsigned int*)take(fpc(32, 64) * 4);
  unsigned int* f_h1   = (unsigned int*)take(fpm(64, 96) * 4);
  unsigned int* f_h2   = (unsigned int*)take(fpm(64, 64) * 4);

  _Float16* a0   = b64b;                         // 16ch, dead before res c1
  _Float16* pong = b64c;                         // 32ch, dead before splat
  float* num = (float*)b64c;                     // 16ch f32, dead before res lateral
  float* den = num + (size_t)BB * 16 * HW;
  _Float16* tmp2 = ping;                         // final latent

  auto blks = [](size_t n) { return (unsigned)((n + 255) / 256); };

  // -------- weight packing
  prep_conv_frags<<<blks(fpc(32, 16)), 256, 0, stream>>>(ec1w, f_ec1, 32, 16);
  for (int l = 0; l < 3; ++l) {
    prep_conv_frags<<<blks(fpc(32, 32)), 256, 0, stream>>>(lw1[l], f_lat[l * 2], 32, 32);
    prep_conv_frags<<<blks(fpc(32, 32)), 256, 0, stream>>>(lw2[l], f_lat[l * 2 + 1], 32, 32);
  }
  prep_conv_frags<<<blks(fpc(16, 32)), 256, 0, stream>>>(ec2w, f_ec2, 16, 32);
  prep_conv_frags<<<blks(fpc(32, 64)), 256, 0, stream>>>(rc0w, f_rc0, 32, 64);
  prep_conv_frags<<<blks(fpc(64, 32)), 256, 0, stream>>>(rc1w, f_rc1, 64, 32);
  prep_conv_frags<<<blks(fpc(64, 64)), 256, 0, stream>>>(rlw1, f_rl1, 64, 64);
  prep_conv_frags<<<blks(fpc(64, 64)), 256, 0, stream>>>(rlw2, f_rl2, 64, 64);
  prep_conv_frags<<<blks(fpc(32, 64)), 256, 0, stream>>>(rc2w, f_rc2, 32, 64);
  prep_mlp_frags<<<blks(fpm(64, 96)), 256, 0, stream>>>(h0w, f_h1, 80, 64, 96);
  prep_mlp_frags<<<blks(fpm(64, 64)), 256, 0, stream>>>(h1w, f_h2, 64, 64, 64);

  // -------- splatting weights (blur/variance/warp/error)
  splat_weights_kernel<<<blks((size_t)BB * HW), 256, 0, stream>>>(oflow, g, a_fe, a_v, w1, w2);

  dim3 cg(WW / 128, HH, BB);
  // -------- encoder per frame
  for (int f = 0; f < 2; ++f) {
    enc_c0_kernel<<<blks((size_t)BB * HW), 256, 0, stream>>>(xs, f, ec0w, ec0b, a0);
    conv3x3_wmma<16, 32, 4><<<cg, 256, 0, stream>>>(a0, 16, f_ec1, ec1b, ping, 32, nullptr, 0, 0, 1, 0);
    for (int l = 0; l < 3; ++l) {
      conv3x3_wmma<32, 32, 3><<<cg, 256, 0, stream>>>(ping, 32, f_lat[l * 2], lb1[l], pong, 32, nullptr, 0, 0, 1, 0);
      conv3x3_wmma<32, 32, 3><<<cg, 256, 0, stream>>>(pong, 32, f_lat[l * 2 + 1], lb2[l], ping, 32, ping, 32, 0, 0, 0);
    }
    conv3x3_wmma<32, 16, 3><<<cg, 256, 0, stream>>>(ping, 32, f_ec2, ec2b,
        act64 + (size_t)f * 16 * HW, 64, nullptr, 0, 1, 0, 1);
  }

  // -------- softmax forward splatting (s0 -> act64 ch32, s1 -> ch48)
  for (int s = 0; s < 2; ++s) {
    zero_f32<<<blks((size_t)BB * 17 * HW), 256, 0, stream>>>(num, BB * 17 * HW);
    splat_kernel<<<blks((size_t)BB * HW), 256, 0, stream>>>(
        act64 + (size_t)s * 16 * HW, 64, oflow, s, tsteps, s == 0 ? w1 : w2, num, den);
    normalize_kernel<<<blks((size_t)BB * 16 * HW), 256, 0, stream>>>(
        num, den, act64 + (size_t)(32 + s * 16) * HW, 64);
  }

  // -------- residual conv block on [pl0|pl1|s0|s1]
  conv3x3_wmma<64, 32, 3><<<cg, 256, 0, stream>>>(act64, 64, f_rc0, rc0b, ping, 32, nullptr, 0, 0, 0, 0);
  conv3x3_wmma<32, 64, 3><<<cg, 256, 0, stream>>>(ping, 32, f_rc1, rc1b, b64b, 64, nullptr, 0, 0, 1, 0);
  conv3x3_wmma<64, 64, 3><<<cg, 256, 0, stream>>>(b64b, 64, f_rl1, rlb1, b64c, 64, nullptr, 0, 0, 1, 0);
  conv3x3_wmma<64, 64, 3><<<cg, 256, 0, stream>>>(b64c, 64, f_rl2, rlb2, b64b, 64, b64b, 64, 0, 0, 0);
  conv3x3_wmma<64, 32, 3><<<cg, 256, 0, stream>>>(b64b, 64, f_rc2, rc2b, tmp2, 32,
      act64 + (size_t)32 * HW, 64, 1, 0, 1);

  // -------- hyponet MLP
  hyponet_kernel<<<cg, 256, 0, stream>>>(coord, tmp2, f_h1, h0b, f_h2, h1b, h2w, h2b, (float*)d_out);
}
